// SimCLRLoss_60773787238637
// MI455X (gfx1250) — matrix operand order repaired and checked by reading, not compiled, not run on hardware
//
#include <hip/hip_runtime.h>

// ---- CDNA5 / gfx1250 vector types (wave32) ----
typedef _Float16 v8h  __attribute__((ext_vector_type(8)));
typedef _Float16 v16h __attribute__((ext_vector_type(16)));
typedef float    v8f  __attribute__((ext_vector_type(8)));
typedef float    v4f  __attribute__((ext_vector_type(4)));

#define DDIM   256      // feature dim
#define BATCH  4096     // B
#define NROWS  8192     // N = 2B
#define NTILES (NROWS / 16)   // 512 column tiles
#define JSPLIT 4              // column-range splits per M-block
#define JT_PER (NTILES / JSPLIT)
// logits = 2*cos_sim in [-2, 2] -> fixed softmax max 2.0 is numerically safe,
// and partial exp-sums over column ranges combine additively.
#define LOGIT_MAX 2.0f

// -------------------------------------------------------------------------
// Kernel 1: row L2-normalize [z_i; z_j] -> fp16 zn[N, D]; zero accumulators.
// -------------------------------------------------------------------------
__global__ __launch_bounds__(256) void simclr_normalize(
    const float* __restrict__ zi, const float* __restrict__ zj,
    _Float16* __restrict__ zn, float* __restrict__ rowsum,
    float* __restrict__ out)
{
    const int tid = blockIdx.x * 256 + threadIdx.x;
    if (tid <= NROWS) rowsum[tid] = 0.0f;            // rowsum[0..8191] + posSum
    if (tid == 0) out[0] = 0.0f;

    const int wave = threadIdx.x >> 5;
    const int lane = threadIdx.x & 31;
    const int row  = blockIdx.x * 8 + wave;          // grid = 1024 blocks

    const float* src = (row < BATCH) ? (zi + (size_t)row * DDIM)
                                     : (zj + (size_t)(row - BATCH) * DDIM);
    v4f a = *(const v4f*)(src + lane * 8);
    v4f b = *(const v4f*)(src + lane * 8 + 4);

    float ss = 0.0f;
    #pragma unroll
    for (int i = 0; i < 4; ++i) ss += a[i] * a[i] + b[i] * b[i];
    #pragma unroll
    for (int off = 16; off; off >>= 1) ss += __shfl_xor(ss, off, 32);

    const float scale = 1.0f / fmaxf(sqrtf(ss), 1e-8f);

    v8h o;
    #pragma unroll
    for (int i = 0; i < 4; ++i) {
        o[i]     = (_Float16)(a[i] * scale);
        o[i + 4] = (_Float16)(b[i] * scale);
    }
    *(v8h*)(zn + (size_t)row * DDIM + lane * 8) = o;
}

// -------------------------------------------------------------------------
// Fragment loaders for v_wmma_f32_16x16x32_f16 (wave32 layouts, ISA 7.12.2).
// A (16x32, 16-bit):  lanes 0-15  row M=lane    : VGPR0-3 K=kb..+7,  VGPR4-7 K=kb+16..+23
//                     lanes 16-31 row M=lane-16 : VGPR0-3 K=kb+8..+15, VGPR4-7 K=kb+24..+31
// B (32x16, 16-bit):  lanes 0-15  col N=lane    : VGPR0-3 K=kb..+7,  VGPR4-7 K=kb+16..+23
//                     lanes 16-31 col N=lane-16 : VGPR0-3 K=kb+8..+15, VGPR4-7 K=kb+24..+31
// Both map to contiguous 16-byte chunks of row-major zn -> global_load_b128.
// -------------------------------------------------------------------------
__device__ __forceinline__ v16h load_a_frag(const _Float16* base /* row + hgrp*8 */, int kb)
{
    v8h lo = *(const v8h*)(base + kb);
    v8h hi = *(const v8h*)(base + kb + 16);
    v16h f;
    #pragma unroll
    for (int i = 0; i < 8; ++i) { f[i] = lo[i]; f[i + 8] = hi[i]; }
    return f;
}

__device__ __forceinline__ v16h load_b_frag(const _Float16* base /* row + hgrp*8 */, int kb)
{
    v8h lo = *(const v8h*)(base + kb);
    v8h hi = *(const v8h*)(base + kb + 16);
    v16h f;
    #pragma unroll
    for (int i = 0; i < 8; ++i) { f[i] = lo[i]; f[i + 8] = hi[i]; }
    return f;
}

// -------------------------------------------------------------------------
// Kernel 2: fused  S = (zn @ zn^T)  +  fixed-max exp-sum over j != i.
// Each wave: one 32-row M-block (two 16-row WMMA tiles sharing every B
// fragment) x one of JSPLIT column ranges (128 tiles).  1024 waves total.
//
// A sched_barrier pins the whole B-tile load batch (16 x b128, 64 VGPRs
// live) ahead of the 16-WMMA chain so waits become partial and loads
// overlap MMAs.  Diagonal / partner columns live in wave-UNIFORM tiles ->
// scalar-branch fixups taken on <=4 of 128 iterations; hot loop has no
// per-element index tests.
// C/D layout: VGPR r, lanes 0-15 -> (M=r, N=lane); lanes 16-31 -> (M=8+r).
// -------------------------------------------------------------------------
__global__ __launch_bounds__(128) void simclr_partial(
    const _Float16* __restrict__ zn, float* __restrict__ rowsum,
    float* __restrict__ posSum)
{
    const int lane  = threadIdx.x & 31;
    // force wave id scalar so tile bounds / fixup ids become SGPRs
    const int wave  = __builtin_amdgcn_readfirstlane(threadIdx.x >> 5);
    const int w     = blockIdx.x * 4 + wave;         // grid = 256 blocks
    const int Mbase = (w >> 2) * 32;                 // 256 M-blocks
    const int jt0   = (w & 3) * JT_PER;              // 4 column ranges
    const int jtEnd = jt0 + JT_PER;
    const int hgrp  = lane >> 4;
    const int l15   = lane & 15;

    // wave-uniform special tiles (SGPR compares in the loop)
    const int jtD0 = Mbase >> 4;                     // diag tile, rows +0..15
    const int jtD1 = jtD0 + 1;                       // diag tile, rows +16..31
    const int pOfs = (Mbase < BATCH) ? (BATCH / 16) : -(BATCH / 16);
    const int jtP0 = jtD0 + pOfs;                    // partner tiles
    const int jtP1 = jtD1 + pOfs;
    // per-lane row selector: lane matches accumulator row r iff rMatch == r
    const int rMatch = l15 - 8 * hgrp;               // in [0,8) on one lane per row

    // ---- preload both A fragment sets (reused for all 128 tiles) ----
    v16h afr0[8], afr1[8];
    {
        const _Float16* arow0 = zn + (size_t)(Mbase + l15) * DDIM + hgrp * 8;
        const _Float16* arow1 = zn + (size_t)(Mbase + 16 + l15) * DDIM + hgrp * 8;
        #pragma unroll
        for (int kk = 0; kk < 8; ++kk) {
            afr0[kk] = load_a_frag(arow0, kk * 32);
            afr1[kk] = load_a_frag(arow1, kk * 32);
        }
    }

    float sum0[8], sum1[8], posAcc = 0.0f;
    #pragma unroll
    for (int r = 0; r < 8; ++r) { sum0[r] = 0.0f; sum1[r] = 0.0f; }

    for (int jt = jt0; jt < jtEnd; ++jt) {
        const _Float16* brow = zn + (size_t)(jt * 16 + l15) * DDIM + hgrp * 8;

        // prefetch next tile's B rows into near caches (global_prefetch_b8)
        __builtin_prefetch(brow + DDIM * 16, 0, 3);

        // ---- issue the full B tile as one load batch ----
        v16h bfr[8];
        #pragma unroll
        for (int kk = 0; kk < 8; ++kk) bfr[kk] = load_b_frag(brow, kk * 32);

        // pin the batch ahead of the MMA chain: nothing may cross ->
        // all 8 fragments stay live, waits become partial/decrementing
        __builtin_amdgcn_sched_barrier(0);

        v8f acc0 = {};
        v8f acc1 = {};
        #pragma unroll
        for (int kk = 0; kk < 8; ++kk) {
            // (neg_a, A, neg_b, B, c_mod, C, reuse_a, reuse_b)
            acc0 = __builtin_amdgcn_wmma_f32_16x16x32_f16(
                false, afr0[kk], false, bfr[kk], (short)0, acc0, false, false);
            acc1 = __builtin_amdgcn_wmma_f32_16x16x32_f16(
                false, afr1[kk], false, bfr[kk], (short)0, acc1, false, false);
        }

        // hot path: unconditional exp-sum (fixed max 2.0; e in [e^-4, 1])
        #pragma unroll
        for (int r = 0; r < 8; ++r) {
            sum0[r] += __expf(fmaf(2.0f, acc0[r], -LOGIT_MAX));
            sum1[r] += __expf(fmaf(2.0f, acc1[r], -LOGIT_MAX));
        }

        // rare fixups: wave-uniform scalar branches (<=4 taken per wave)
        if (jt == jtD0) {        // remove self-diagonal terms, rows +0..15
            #pragma unroll
            for (int r = 0; r < 8; ++r)
                sum0[r] -= (rMatch == r) ? __expf(fmaf(2.0f, acc0[r], -LOGIT_MAX)) : 0.0f;
        }
        if (jt == jtD1) {        // remove self-diagonal terms, rows +16..31
            #pragma unroll
            for (int r = 0; r < 8; ++r)
                sum1[r] -= (rMatch == r) ? __expf(fmaf(2.0f, acc1[r], -LOGIT_MAX)) : 0.0f;
        }
        if (jt == jtP0) {        // harvest positive similarities, rows +0..15
            #pragma unroll
            for (int r = 0; r < 8; ++r)
                posAcc += (rMatch == r) ? acc0[r] : 0.0f;
        }
        if (jt == jtP1) {        // harvest positive similarities, rows +16..31
            #pragma unroll
            for (int r = 0; r < 8; ++r)
                posAcc += (rMatch == r) ? acc1[r] : 0.0f;
        }
    }

    // ---- per-row partial sums: reduce across 16 lanes of each half-group ----
    #pragma unroll
    for (int r = 0; r < 8; ++r) {
        float s0 = sum0[r], s1 = sum1[r];
        #pragma unroll
        for (int off = 8; off; off >>= 1) {
            s0 += __shfl_xor(s0, off, 16);
            s1 += __shfl_xor(s1, off, 16);
        }
        if (l15 == 0) {                                  // lanes 0 and 16
            const int row0 = Mbase + r + hgrp * 8;
            atomicAdd(&rowsum[row0],      s0);
            atomicAdd(&rowsum[row0 + 16], s1);
        }
    }

    // ---- positive-similarity total: reduce across the full wave ----
    #pragma unroll
    for (int off = 16; off; off >>= 1) posAcc += __shfl_xor(posAcc, off, 32);
    if (lane == 0) atomicAdd(posSum, posAcc);
}

// -------------------------------------------------------------------------
// Kernel 3: loss = [ sum_i (2 + log(rowsum_i)) - 2 * posSum ] / N
// -------------------------------------------------------------------------
__global__ __launch_bounds__(256) void simclr_finalize(
    const float* __restrict__ rowsum, const float* __restrict__ posSum,
    float* __restrict__ out)
{
    const int i = blockIdx.x * 256 + threadIdx.x;    // grid = 32 blocks
    float t = LOGIT_MAX + logf(rowsum[i]);
    #pragma unroll
    for (int off = 16; off; off >>= 1) t += __shfl_xor(t, off, 32);
    if ((threadIdx.x & 31) == 0) atomicAdd(out, t * (1.0f / (float)NROWS));
    if (i == 0) atomicAdd(out, -2.0f * posSum[0] * (1.0f / (float)NROWS));
}

// -------------------------------------------------------------------------
extern "C" void kernel_launch(void* const* d_in, const int* in_sizes, int n_in,
                              void* d_out, int out_size, void* d_ws, size_t ws_size,
                              hipStream_t stream)
{
    (void)in_sizes; (void)n_in; (void)out_size; (void)ws_size;
    const float* zi  = (const float*)d_in[0];
    const float* zj  = (const float*)d_in[1];
    float*       out = (float*)d_out;

    _Float16* zn     = (_Float16*)d_ws;                            // 4 MB fp16
    float*    rowsum = (float*)((char*)d_ws + (size_t)NROWS * DDIM * 2);
    float*    posSum = rowsum + NROWS;                             // 1 scalar

    simclr_normalize<<<NROWS / 8, 256, 0, stream>>>(zi, zj, zn, rowsum, out);
    simclr_partial<<<(NROWS / 32) * JSPLIT / 4, 128, 0, stream>>>(zn, rowsum, posSum);
    simclr_finalize<<<NROWS / 256, 256, 0, stream>>>(rowsum, posSum, out);
}
